// CellTrack_Model_58256936403402
// MI455X (gfx1250) — compile-verified, matrix operand order checked by reading, not run on hardware
//
#include <hip/hip_runtime.h>
#include <hip/hip_bf16.h>
#include <hip/hip_fp16.h>

typedef __attribute__((ext_vector_type(16))) _Float16 v16h;
typedef __attribute__((ext_vector_type(8)))  float    v8f;

#define NN   50000
#define EE   500000
#define DHH  13
#define DLL  128
#define DD   128
#define HHID 256
#define HCC  64
#define EINW 527
#define EINP 544   // 527 padded to 17*32

// ---------------- WMMA fragment layout helpers (gfx1250, wave32) -----------
// A 16x32 f16: lane L -> row m=L&15, koff=(L&16)?8:0; half j -> k = koff + (j<8?j:j+8)
// Fragment storage: k-tile kt occupies 512 halfs; lane L's 16 halfs contiguous.
// frag_idx inverts (m,k) -> storage index.
__device__ __forceinline__ int frag_idx(int m, int k) {
  int kk = k & 31;
  int lane = m + ((kk & 8) ? 16 : 0);
  int j = (kk & 7) + ((kk & 16) ? 8 : 0);
  return ((k >> 5) << 9) + (lane << 4) + j;
}

__device__ __forceinline__ v8f wmma_step(v16h a, v16h b, v8f c) {
  return __builtin_amdgcn_wmma_f32_16x16x32_f16(false, a, false, b, (short)0, c,
                                                false, false);
}

// A operand: contiguous 32B per lane from LDS fragment buffer
__device__ __forceinline__ v16h load_a_frag(const _Float16* s, int kt, int lane) {
  return *(const v16h*)(s + (kt << 9) + (lane << 4));
}

// B operand: contiguous 32B per lane from pre-packed global weights.
// Tile order: tile = nt * nkt + kt, each tile 512 halfs.
__device__ __forceinline__ v16h load_b_p(const _Float16* __restrict__ P, int nkt,
                                         int nt, int kt, int lane) {
  return *(const v16h*)(P + (((size_t)(nt * nkt + kt)) << 9) + (lane << 4));
}

__device__ __forceinline__ v8f bias_acc(const float* __restrict__ bias, int n0, int lane) {
  float bv = bias[n0 + (lane & 15)];
  v8f c;
#pragma unroll
  for (int i = 0; i < 8; ++i) c[i] = bv;
  return c;
}

// C/D: lane L, VGPR i -> (m = i + (L>=16?8:0), n = L%16)
__device__ __forceinline__ void store_c_frag_f16(_Float16* s, int n0, v8f c,
                                                 int lane, bool relu) {
  const int mbase = (lane & 16) ? 8 : 0;
  const int n = n0 + (lane & 15);
#pragma unroll
  for (int i = 0; i < 8; ++i) {
    float v = c[i];
    if (relu) v = fmaxf(v, 0.0f);
    s[frag_idx(mbase + i, n)] = (_Float16)v;
  }
}

__device__ __forceinline__ void store_c_lds_f32(float* s, int lda, int n0,
                                                v8f c, int lane, bool relu) {
  const int mbase = (lane & 16) ? 8 : 0;
  const int n = n0 + (lane & 15);
#pragma unroll
  for (int i = 0; i < 8; ++i) {
    float v = c[i];
    if (relu) v = fmaxf(v, 0.0f);
    s[(mbase + i) * lda + n] = v;
  }
}

// ---------------- K-pack: f32 row-major weights -> f16 B-fragment tiles ----
__global__ void k_pack_w(const float* __restrict__ W, _Float16* __restrict__ P,
                         int K, int Nout, int nkt, int total) {
  int idx = blockIdx.x * blockDim.x + threadIdx.x;
  if (idx >= total) return;
  int lane = idx & 31;
  int tile = idx >> 5;
  int nt = tile / nkt, kt = tile - nt * nkt;
  int n = nt * 16 + (lane & 15);
  int koff = (lane & 16) ? 8 : 0;
  _Float16* dst = P + ((size_t)tile << 9) + (lane << 4);
#pragma unroll
  for (int j = 0; j < 16; ++j) {
    int k = kt * 32 + koff + ((j < 8) ? j : (j + 8));
    float w = (k < K) ? W[(size_t)k * Nout + n] : 0.0f;
    dst[j] = (_Float16)w;
  }
}

// ---------------- K0: zero the aggregation buffer ----------------
__global__ void k_zero(float* __restrict__ p, int n) {
  int i = blockIdx.x * blockDim.x + threadIdx.x;
  if (i < n) p[i] = 0.0f;
}

// ---------------- K1: node encoders -> x = [h1|h2] ----------------
// x16 out: fragment-tile layout, 2048 halfs per 16-node tile (k-tiles 0..3)
__global__ __launch_bounds__(256) void k_node_encode(
    const float* __restrict__ x1, const float* __restrict__ x2,
    const _Float16* __restrict__ Pwh1, const float* __restrict__ bh1,
    const _Float16* __restrict__ Pwh2, const float* __restrict__ bh2,
    const _Float16* __restrict__ Pwl1, const float* __restrict__ bl1,
    const _Float16* __restrict__ Pwl2, const float* __restrict__ bl2,
    float* __restrict__ x32, _Float16* __restrict__ x16) {
  __shared__ __align__(16) _Float16 sX1[16 * 32];
  __shared__ __align__(16) _Float16 sX2[16 * 128];
  __shared__ __align__(16) _Float16 sH1[16 * 256];
  __shared__ __align__(16) _Float16 sH2[16 * 256];
  const int tid = threadIdx.x, lane = tid & 31, wave = tid >> 5;
  const int t0 = blockIdx.x * 16;

  for (int i = tid; i < 16 * 32; i += 256) {
    int r = i >> 5, c = i & 31;
    sX1[frag_idx(r, c)] =
        (_Float16)((c < DHH) ? x1[(size_t)(t0 + r) * DHH + c] : 0.0f);
  }
  for (int i = tid; i < 16 * 128; i += 256) {
    int r = i >> 7, c = i & 127;
    sX2[frag_idx(r, c)] = (_Float16)x2[(size_t)(t0 + r) * DLL + c];
  }
  __syncthreads();

  // Layer 1 hidden for both encoders (H=256 -> 16 n-tiles, 2 per wave)
  for (int nt = wave; nt < 16; nt += 8) {
    int n0 = nt * 16;
    v8f c1 = bias_acc(bh1, n0, lane);
    c1 = wmma_step(load_a_frag(sX1, 0, lane), load_b_p(Pwh1, 1, nt, 0, lane), c1);
    store_c_frag_f16(sH1, n0, c1, lane, true);

    v8f c2 = bias_acc(bl1, n0, lane);
#pragma unroll
    for (int ks = 0; ks < 4; ++ks)
      c2 = wmma_step(load_a_frag(sX2, ks, lane), load_b_p(Pwl1, 4, nt, ks, lane), c2);
    store_c_frag_f16(sH2, n0, c2, lane, true);
  }
  __syncthreads();

  // Layer 2: waves 0-1 -> h1 out (32 cols); waves 2-7 -> h2 out (96 cols)
  v8f c;
  int gcol0;
  if (wave < 2) {
    int n0 = wave * 16;
    gcol0 = n0;
    c = bias_acc(bh2, n0, lane);
#pragma unroll
    for (int ks = 0; ks < 8; ++ks)
      c = wmma_step(load_a_frag(sH1, ks, lane), load_b_p(Pwh2, 8, wave, ks, lane), c);
  } else {
    int n0 = (wave - 2) * 16;
    gcol0 = 32 + n0;
    c = bias_acc(bl2, n0, lane);
#pragma unroll
    for (int ks = 0; ks < 8; ++ks)
      c = wmma_step(load_a_frag(sH2, ks, lane), load_b_p(Pwl2, 8, wave - 2, ks, lane), c);
  }
  {
    const int mbase = (lane & 16) ? 8 : 0;
    const int nc = gcol0 + (lane & 15);
    const size_t tb = (size_t)blockIdx.x * 2048;
#pragma unroll
    for (int i = 0; i < 8; ++i) {
      int m = mbase + i;
      float v = c[i];
      x32[(size_t)(t0 + m) * DD + nc] = v;
      x16[tb + frag_idx(m, nc)] = (_Float16)v;
    }
  }
}

// ---------------- K2: fused edge encoder + msg scatter ----------------
// e16 out: fragment-tile layout, 2048 halfs per 16-edge tile
__global__ __launch_bounds__(256) void k_edge_encode(
    const float* __restrict__ x1, const float* __restrict__ x2,
    const int* __restrict__ ei, const float* __restrict__ x32,
    const _Float16* __restrict__ Pwe1, const float* __restrict__ be1,
    const _Float16* __restrict__ Pwe2, const float* __restrict__ be2,
    const _Float16* __restrict__ Pwmsg, const float* __restrict__ bmsg,
    _Float16* __restrict__ e16, float* __restrict__ agg) {
  __shared__ int s_src[16], s_trg[16];
  __shared__ __align__(16) float sXi[2 * 16 * 144];   // raw gathers (aliased)
  __shared__ __align__(16) float sXf[2 * 16 * 128];   // encoder-output gathers
  __shared__ __align__(16) _Float16 sFeat[16 * EINP]; // frag, 17 k-tiles
  __shared__ __align__(16) _Float16 sHid[16 * 256];   // frag, 8 k-tiles
  _Float16* sCat = (_Float16*)sXi;                    // frag [xs|e], aliases sXi
  const int tid = threadIdx.x, lane = tid & 31, wave = tid >> 5;
  const int e0 = blockIdx.x * 16;

  if (tid < 16) {
    s_src[tid] = ei[e0 + tid];
    s_trg[tid] = ei[EE + e0 + tid];
  }
  __syncthreads();

  for (int i = tid; i < 2 * 16 * 144; i += 256) {
    int which = i / (16 * 144), rem = i % (16 * 144);
    int r = rem / 144, c = rem % 144;
    int node = which ? s_trg[r] : s_src[r];
    float v = 0.0f;
    if (c < DHH) v = x1[(size_t)node * DHH + c];
    else if (c < 141) v = x2[(size_t)node * DLL + (c - DHH)];
    sXi[i] = v;
  }
  for (int i = tid; i < 2 * 16 * 128; i += 256) {
    int which = i >> 11, rem = i & 2047;
    int r = rem >> 7, c = rem & 127;
    int node = which ? s_trg[r] : s_src[r];
    sXf[i] = x32[(size_t)node * DD + c];
  }
  __syncthreads();

  // cosine sims (threads 0-15: sim1 over 141; threads 16-31: sim2 over 128)
  if (tid < 16) {
    float dot = 0.f, na = 0.f, nb = 0.f;
    for (int c = 0; c < 141; ++c) {
      float a = sXi[tid * 144 + c], b = sXi[16 * 144 + tid * 144 + c];
      dot += a * b; na += a * a; nb += b * b;
    }
    float d = fmaxf(sqrtf(na), 1e-8f) * fmaxf(sqrtf(nb), 1e-8f);
    sFeat[frag_idx(tid, 141)] = (_Float16)(dot / d);
  } else if (tid < 32) {
    int r = tid - 16;
    float dot = 0.f, na = 0.f, nb = 0.f;
    for (int c = 0; c < 128; ++c) {
      float a = sXf[r * 128 + c], b = sXf[16 * 128 + r * 128 + c];
      dot += a * b; na += a * a; nb += b * b;
    }
    float d = fmaxf(sqrtf(na), 1e-8f) * fmaxf(sqrtf(nb), 1e-8f);
    sFeat[frag_idx(r, 526)] = (_Float16)(dot / d);
  }
  // assemble [abs_init | sim1 | xs | xt | |xs-xt| | sim2 | pad] (frag layout)
  for (int i = tid; i < 16 * EINP; i += 256) {
    int r = i / EINP, c = i % EINP;
    if (c == 141 || c == 526) continue;
    float v = 0.0f;
    if (c < 141)      v = fabsf(sXi[r * 144 + c] - sXi[16 * 144 + r * 144 + c]);
    else if (c < 270) v = sXf[r * 128 + (c - 142)];
    else if (c < 398) v = sXf[16 * 128 + r * 128 + (c - 270)];
    else if (c < 526) v = fabsf(sXf[r * 128 + (c - 398)] -
                                sXf[16 * 128 + r * 128 + (c - 398)]);
    sFeat[frag_idx(r, c)] = (_Float16)v;
  }
  __syncthreads();

  // sCat k-tiles 0..3 = xs (sXi now dead; sCat aliases it)
  for (int i = tid; i < 16 * 128; i += 256) {
    int r = i >> 7, c = i & 127;
    sCat[frag_idx(r, c)] = (_Float16)sXf[r * 128 + c];
  }
  __syncthreads();

  // MM1: hidden = relu(feat @ We1 + be1), K=544 (zero-padded), N=256
  for (int nt = wave; nt < 16; nt += 8) {
    int n0 = nt * 16;
    v8f c = bias_acc(be1, n0, lane);
#pragma unroll 1
    for (int ks = 0; ks < 17; ++ks)
      c = wmma_step(load_a_frag(sFeat, ks, lane), load_b_p(Pwe1, 17, nt, ks, lane), c);
    store_c_frag_f16(sHid, n0, c, lane, true);
  }
  __syncthreads();

  // MM2: e = hidden @ We2 + be2 (linear), K=256, N=128
  {
    int n0 = wave * 16;
    v8f c = bias_acc(be2, n0, lane);
#pragma unroll
    for (int ks = 0; ks < 8; ++ks)
      c = wmma_step(load_a_frag(sHid, ks, lane), load_b_p(Pwe2, 8, wave, ks, lane), c);
    const int mbase = (lane & 16) ? 8 : 0;
    const int n = n0 + (lane & 15);
    const size_t eb = (size_t)blockIdx.x * 2048;
#pragma unroll
    for (int i = 0; i < 8; ++i) {
      int m = mbase + i;
      _Float16 h = (_Float16)c[i];
      int fi = frag_idx(m, n);
      e16[eb + fi] = h;
      sCat[(4 << 9) + fi] = h;  // k-tiles 4..7 of sCat
    }
  }
  __syncthreads();

  // MM3: msg = relu([xs|e] @ Wmsg + bmsg); scatter-add into agg[trg]
  {
    int n0 = wave * 16;
    v8f c = bias_acc(bmsg, n0, lane);
#pragma unroll
    for (int ks = 0; ks < 8; ++ks)
      c = wmma_step(load_a_frag(sCat, ks, lane), load_b_p(Pwmsg, 8, wave, ks, lane), c);
    const int mbase = (lane & 16) ? 8 : 0;
    const int n = n0 + (lane & 15);
#pragma unroll
    for (int i = 0; i < 8; ++i) {
      float v = fmaxf(c[i], 0.0f);
      int node = s_trg[mbase + i];
      unsafeAtomicAdd(&agg[(size_t)node * DD + n], v);  // global_atomic_add_f32
    }
  }
}

// ---------------- K3: node update xn = relu([x|agg] @ Wnode + b) ------------
__global__ __launch_bounds__(256) void k_node_update(
    const _Float16* __restrict__ x16, const float* __restrict__ agg,
    const _Float16* __restrict__ Pwnode, const float* __restrict__ bnode,
    _Float16* __restrict__ xn16) {
  __shared__ __align__(16) _Float16 sA[16 * 256];  // frag, 8 k-tiles
  const int tid = threadIdx.x, lane = tid & 31, wave = tid >> 5;
  const int t0 = blockIdx.x * 16;

  // k-tiles 0..3: x16 tile already in fragment layout -> straight copy
  {
    const uint4* src = (const uint4*)(x16 + (size_t)blockIdx.x * 2048);
    ((uint4*)sA)[tid] = src[tid];  // 256 x 16B = 4096B = 2048 halfs
  }
  // k-tiles 4..7: agg (f32, coalesced read; scattered 2B frag writes)
  for (int i = tid; i < 16 * 128; i += 256) {
    int r = i >> 7, c = i & 127;
    sA[frag_idx(r, 128 + c)] = (_Float16)agg[(size_t)(t0 + r) * DD + c];
  }
  __syncthreads();

  int n0 = wave * 16;
  v8f c = bias_acc(bnode, n0, lane);
#pragma unroll
  for (int ks = 0; ks < 8; ++ks)
    c = wmma_step(load_a_frag(sA, ks, lane), load_b_p(Pwnode, 8, wave, ks, lane), c);
  const int mbase = (lane & 16) ? 8 : 0;
  const int n = n0 + (lane & 15);
#pragma unroll
  for (int i = 0; i < 8; ++i) {
    float v = fmaxf(c[i], 0.0f);
    xn16[(size_t)(t0 + mbase + i) * DD + n] = (_Float16)v;  // row-major (gathered later)
  }
}

// ---------------- K4: final edge MLP + classifier ----------------
__global__ __launch_bounds__(256) void k_edge_final(
    const int* __restrict__ ei, const _Float16* __restrict__ xn16,
    const _Float16* __restrict__ e16,
    const _Float16* __restrict__ Pwmp1, const float* __restrict__ bmp1,
    const _Float16* __restrict__ Pwmp2, const float* __restrict__ bmp2,
    const _Float16* __restrict__ Pwc1, const float* __restrict__ bc1,
    const float* __restrict__ Wc2, const float* __restrict__ bc2,
    float* __restrict__ out) {
  __shared__ int s_src[16], s_trg[16];
  __shared__ __align__(16) _Float16 sA[16 * 384];    // frag, 12 k-tiles
  __shared__ __align__(16) _Float16 sHid[16 * 256];  // frag, 8 k-tiles
  __shared__ __align__(16) _Float16 sEmp[16 * 128];  // frag, 4 k-tiles
  __shared__ float sHc[16 * 64];
  const int tid = threadIdx.x, lane = tid & 31, wave = tid >> 5;
  const int e0 = blockIdx.x * 16;

  if (tid < 16) {
    s_src[tid] = ei[e0 + tid];
    s_trg[tid] = ei[EE + e0 + tid];
  }
  // k-tiles 8..11: e16 tile already in fragment layout -> straight copy
  {
    const uint4* esrc = (const uint4*)(e16 + (size_t)blockIdx.x * 2048);
    ((uint4*)(sA + (8 << 9)))[tid] = esrc[tid];  // 256 x 16B
  }
  __syncthreads();
  // k-tiles 0..7: gathered xn[src], xn[trg] rows
  for (int i = tid; i < 16 * 256; i += 256) {
    int r = i >> 8, c = i & 255;
    int node = (c < 128) ? s_src[r] : s_trg[r];
    sA[frag_idx(r, c)] = xn16[(size_t)node * DD + (c & 127)];
  }
  __syncthreads();

  // hid = relu([xn_s|xn_t|e] @ Wmp1 + bmp1), K=384, N=256
  for (int nt = wave; nt < 16; nt += 8) {
    int n0 = nt * 16;
    v8f c = bias_acc(bmp1, n0, lane);
#pragma unroll 1
    for (int ks = 0; ks < 12; ++ks)
      c = wmma_step(load_a_frag(sA, ks, lane), load_b_p(Pwmp1, 12, nt, ks, lane), c);
    store_c_frag_f16(sHid, n0, c, lane, true);
  }
  __syncthreads();

  // e_mp = hid @ Wmp2 + bmp2 (linear), K=256, N=128
  {
    int n0 = wave * 16;
    v8f c = bias_acc(bmp2, n0, lane);
#pragma unroll
    for (int ks = 0; ks < 8; ++ks)
      c = wmma_step(load_a_frag(sHid, ks, lane), load_b_p(Pwmp2, 8, wave, ks, lane), c);
    store_c_frag_f16(sEmp, n0, c, lane, false);
  }
  __syncthreads();

  // hc = relu(e_mp @ Wc1 + bc1), K=128, N=64 (waves 0-3)
  if (wave < 4) {
    int n0 = wave * 16;
    v8f c = bias_acc(bc1, n0, lane);
#pragma unroll
    for (int ks = 0; ks < 4; ++ks)
      c = wmma_step(load_a_frag(sEmp, ks, lane), load_b_p(Pwc1, 4, wave, ks, lane), c);
    store_c_lds_f32(sHc, 64, n0, c, lane, true);
  }
  __syncthreads();

  if (tid < 16) {
    float acc = bc2[0];
    for (int j = 0; j < 64; ++j) acc += sHc[tid * 64 + j] * Wc2[j];
    out[e0 + tid] = acc;
  }
}

// ---------------- launch ----------------
extern "C" void kernel_launch(void* const* d_in, const int* in_sizes, int n_in,
                              void* d_out, int out_size, void* d_ws, size_t ws_size,
                              hipStream_t stream) {
  const float* x1   = (const float*)d_in[0];
  const float* x2   = (const float*)d_in[1];
  const int*   ei   = (const int*)d_in[2];
  const float* Wh1  = (const float*)d_in[3];  const float* bh1 = (const float*)d_in[4];
  const float* Wh2  = (const float*)d_in[5];  const float* bh2 = (const float*)d_in[6];
  const float* Wl1  = (const float*)d_in[7];  const float* bl1 = (const float*)d_in[8];
  const float* Wl2  = (const float*)d_in[9];  const float* bl2 = (const float*)d_in[10];
  const float* We1  = (const float*)d_in[11]; const float* be1 = (const float*)d_in[12];
  const float* We2  = (const float*)d_in[13]; const float* be2 = (const float*)d_in[14];
  const float* Wmsg = (const float*)d_in[15]; const float* bmsg = (const float*)d_in[16];
  const float* Wnode= (const float*)d_in[17]; const float* bnode= (const float*)d_in[18];
  const float* Wmp1 = (const float*)d_in[19]; const float* bmp1 = (const float*)d_in[20];
  const float* Wmp2 = (const float*)d_in[21]; const float* bmp2 = (const float*)d_in[22];
  const float* Wc1  = (const float*)d_in[23]; const float* bc1  = (const float*)d_in[24];
  const float* Wc2  = (const float*)d_in[25]; const float* bc2  = (const float*)d_in[26];

  char* ws = (char*)d_ws;
  const size_t OFF_X16  = 0;          // 3125 tiles * 2048 halfs = 12.8 MB
  const size_t OFF_X32  = 12800000;   // N*128 f32 = 25.6 MB
  const size_t OFF_E16  = 38400000;   // 31250 tiles * 2048 halfs = 128 MB
  const size_t OFF_AGG  = 166400000;  // N*128 f32 = 25.6 MB
  const size_t OFF_XN16 = 192000000;  // N*128 f16 = 12.8 MB (row-major)
  const size_t OFF_PACK = 204800000;  // packed f16 weights, ~0.9 MB
  _Float16* x16  = (_Float16*)(ws + OFF_X16);
  float*    x32  = (float*)(ws + OFF_X32);
  _Float16* e16  = (_Float16*)(ws + OFF_E16);
  float*    agg  = (float*)(ws + OFF_AGG);
  _Float16* xn16 = (_Float16*)(ws + OFF_XN16);
  _Float16* pk   = (_Float16*)(ws + OFF_PACK);

  // packed-weight sub-offsets (in halfs); each = nkt*ntn*512
  _Float16* Pwh1  = pk + 0;       // nkt=1,  ntn=16 ->   8192
  _Float16* Pwl1  = pk + 8192;    // nkt=4,  ntn=16 ->  32768
  _Float16* Pwh2  = pk + 40960;   // nkt=8,  ntn=2  ->   8192
  _Float16* Pwl2  = pk + 49152;   // nkt=8,  ntn=6  ->  24576
  _Float16* Pwe1  = pk + 73728;   // nkt=17, ntn=16 -> 139264
  _Float16* Pwe2  = pk + 212992;  // nkt=8,  ntn=8  ->  32768
  _Float16* Pwmsg = pk + 245760;  // nkt=8,  ntn=8  ->  32768
  _Float16* Pwnode= pk + 278528;  // nkt=8,  ntn=8  ->  32768
  _Float16* Pwmp1 = pk + 311296;  // nkt=12, ntn=16 ->  98304
  _Float16* Pwmp2 = pk + 409600;  // nkt=8,  ntn=8  ->  32768
  _Float16* Pwc1  = pk + 442368;  // nkt=4,  ntn=4  ->   8192
  float* out = (float*)d_out;

  auto pack = [&](const float* W, _Float16* P, int K, int Nout, int nkt) {
    int total = nkt * (Nout >> 4) * 32;
    k_pack_w<<<(total + 255) / 256, 256, 0, stream>>>(W, P, K, Nout, nkt, total);
  };
  pack(Wh1,  Pwh1,  13,  256, 1);
  pack(Wl1,  Pwl1,  128, 256, 4);
  pack(Wh2,  Pwh2,  256, 32,  8);
  pack(Wl2,  Pwl2,  256, 96,  8);
  pack(We1,  Pwe1,  527, 256, 17);
  pack(We2,  Pwe2,  256, 128, 8);
  pack(Wmsg, Pwmsg, 256, 128, 8);
  pack(Wnode,Pwnode,256, 128, 8);
  pack(Wmp1, Pwmp1, 384, 256, 12);
  pack(Wmp2, Pwmp2, 256, 128, 8);
  pack(Wc1,  Pwc1,  128, 64,  4);

  const int nAgg = NN * DD;
  k_zero<<<(nAgg + 255) / 256, 256, 0, stream>>>(agg, nAgg);
  k_node_encode<<<NN / 16, 256, 0, stream>>>(x1, x2, Pwh1, bh1, Pwh2, bh2,
                                             Pwl1, bl1, Pwl2, bl2, x32, x16);
  k_edge_encode<<<EE / 16, 256, 0, stream>>>(x1, x2, ei, x32, Pwe1, be1, Pwe2, be2,
                                             Pwmsg, bmsg, e16, agg);
  k_node_update<<<NN / 16, 256, 0, stream>>>(x16, agg, Pwnode, bnode, xn16);
  k_edge_final<<<EE / 16, 256, 0, stream>>>(ei, xn16, e16, Pwmp1, bmp1, Pwmp2, bmp2,
                                            Pwc1, bc1, Wc2, bc2, out);
}
